// HyenaConv_57681410785535
// MI455X (gfx1250) — compile-verified, matrix operand order checked by reading, not run on hardware
//
#include <hip/hip_runtime.h>

// ---------------------------------------------------------------------------
// Hyena causal conv (len 2048) as block-Toeplitz GEMM on CDNA5 WMMA (bf16).
//   D[m,n] += A[m,k]*B[k,n],  A[m,k]=ker[lbase+m-k], B[k,n]=x[xbase+16n+k]
//   with lbase+xbase = t0  =>  output t = t0 + m + 16n (k cancels: true GEMM).
// A is a descending Toeplitz gather -> stage the kernel REVERSED in LDS
// (two copies, shifted by 1) so every A pair is one aligned ds_load_b32.
// B is contiguous -> two aligned ds_load_b128 per fragment.
// Manual 2x unroll of the lag loop keeps two WMMAs' operands in flight.
// ---------------------------------------------------------------------------

typedef __attribute__((ext_vector_type(16))) __bf16          v16bf;
typedef __attribute__((ext_vector_type(8)))  float           v8f;
typedef __attribute__((ext_vector_type(8)))  unsigned int    u32x8;
typedef __attribute__((ext_vector_type(4)))  unsigned int    u32x4;

#define SEQ     2048
#define XPAD_F  256                       // deepest lag block reaches x[-256]
#define XPAD_B  48
#define XS_LEN  (XPAD_F + SEQ + XPAD_B)   // 2352 bf16
#define GS      2064                      // reversed kernel: G0[i] = ker[GS-i]
#define G_LEN   2100                      // covers reads [2..2096], zero-padded

__device__ __forceinline__ unsigned short f2bf(float f) {
  unsigned int u = __builtin_bit_cast(unsigned int, f);
  u += 0x7FFFu + ((u >> 16) & 1u);        // round to nearest even
  return (unsigned short)(u >> 16);
}

__global__ __launch_bounds__(256) void hyena_conv_wmma_bf16(
    const float* __restrict__ x, const float* __restrict__ kern,
    const float* __restrict__ bias, float* __restrict__ out) {
  __shared__ alignas(16) unsigned short xs[2][XS_LEN];
  __shared__ alignas(16) unsigned short g0[2][G_LEN];   // ker[GS   - i]
  __shared__ alignas(16) unsigned short g1[2][G_LEN];   // ker[GS+1 - i]

  const int tid = threadIdx.x;

  // ---- stage 2 channels: x as padded bf16; kernel reversed, 2 copies ----
  for (int cl = 0; cl < 2; ++cl) {
    const int ch = blockIdx.x * 2 + cl;
    const float4* xrow = (const float4*)(x + (size_t)ch * SEQ);
    const float*  krow = kern + (size_t)ch * SEQ;
    for (int j = tid; j < SEQ / 4; j += 256) {          // 2 iters / thread
      float4 v = xrow[j];
      ushort4 b; b.x = f2bf(v.x); b.y = f2bf(v.y); b.z = f2bf(v.z); b.w = f2bf(v.w);
      *(ushort4*)&xs[cl][XPAD_F + 4 * j] = b;
    }
    if (tid < XPAD_F) xs[cl][tid] = 0;
    if (tid < XPAD_B) xs[cl][XPAD_F + SEQ + tid] = 0;
    // reversed kernel copies: one global load feeds both shifted copies
    for (int i = tid; i < G_LEN - 2; i += 256) {        // ~9 iters / thread
      const int j = GS - i;                             // (-33 .. 2064]
      const unsigned short v =
          (j >= 0 && j < SEQ) ? f2bf(krow[j]) : (unsigned short)0;
      g0[cl][i]     = v;                                // G0[i]   = ker[GS-i]
      g1[cl][i + 1] = v;                                // G1[i+1] = ker[GS-i]
    }
    if (tid == 0) { g1[cl][0] = 0; g0[cl][G_LEN-2] = 0; g0[cl][G_LEN-1] = 0; }
  }
  __syncthreads();

  const int wave = tid >> 5;         // 0..7 (wave32)
  const int lane = tid & 31;
  const int half = lane >> 4;        // K-half selector per ISA frag layout
  const int l15  = lane & 15;        // A: M row   B: N col   D: N col

  // One WMMA accumulation step: aligned A (8 x b32) + B (2 x b128) from LDS.
  auto step = [&](const unsigned short*& pa, const unsigned short*& pb,
                  v8f& acc) {
    u32x8 av;
    av[0] = *(const unsigned int*)__builtin_assume_aligned(pa + 0,  4);
    av[1] = *(const unsigned int*)__builtin_assume_aligned(pa + 2,  4);
    av[2] = *(const unsigned int*)__builtin_assume_aligned(pa + 4,  4);
    av[3] = *(const unsigned int*)__builtin_assume_aligned(pa + 6,  4);
    av[4] = *(const unsigned int*)__builtin_assume_aligned(pa + 16, 4);
    av[5] = *(const unsigned int*)__builtin_assume_aligned(pa + 18, 4);
    av[6] = *(const unsigned int*)__builtin_assume_aligned(pa + 20, 4);
    av[7] = *(const unsigned int*)__builtin_assume_aligned(pa + 22, 4);
    const u32x4 b0 = *(const u32x4*)__builtin_assume_aligned(pb,      16);
    const u32x4 b1 = *(const u32x4*)__builtin_assume_aligned(pb + 16, 16);
    u32x8 bv;
    bv[0] = b0[0]; bv[1] = b0[1]; bv[2] = b0[2]; bv[3] = b0[3];
    bv[4] = b1[0]; bv[5] = b1[1]; bv[6] = b1[2]; bv[7] = b1[3];
    acc = __builtin_amdgcn_wmma_f32_16x16x32_bf16(
        false, __builtin_bit_cast(v16bf, av),
        false, __builtin_bit_cast(v16bf, bv),
        (short)0, acc, false, false);
    pa -= 32;                                  // next lag block
    pb -= 32;
  };

  // Each wave: tile w of channel 0, tile 7-w of channel 1 -> 74 WMMAs, balanced.
  for (int job = 0; job < 2; ++job) {
    const int cl = job;
    const int w  = job ? (7 - wave) : wave;
    const int ch = blockIdx.x * 2 + cl;
    const int t0 = w << 8;                       // 256 outputs per tile group
    const int pmax = 8 * w + 8;                  // lag blocks of 32

    // Prefetch the epilogue residual region while we crunch WMMAs.
    const int tb = t0 + 8 * half + 16 * l15;     // 8 consecutive outputs
    __builtin_prefetch(x + (size_t)ch * SEQ + tb, 0, 3);

    const unsigned short* gk = (l15 & 1) ? g1[cl] : g0[cl];
    const unsigned short* pa = gk + (GS + (l15 & 1) - l15 + 8 * half);
    const unsigned short* pb = &xs[cl][XPAD_F + t0 + 16 * l15 + 8 * half];
    v8f acc = {};

    step(pa, pb, acc);                           // p = 0 (trip count is odd)
    for (int p = 1; p <= pmax; p += 2) {         // pairs: 2 WMMAs in flight
      step(pa, pb, acc);
      step(pa, pb, acc);
    }

    // D layout: acc[r] -> M = r + 8*half, N = l15  =>  t = t0 + M + 16*N
    const float bv = bias[ch & (2048 - 1)];
    const float4* xf = (const float4*)(x + (size_t)ch * SEQ + tb);
    float4 x0 = xf[0], x1 = xf[1];
    float4 o0, o1;
    o0.x = acc[0] + x0.x + bv;  o0.y = acc[1] + x0.y + bv;
    o0.z = acc[2] + x0.z + bv;  o0.w = acc[3] + x0.w + bv;
    o1.x = acc[4] + x1.x + bv;  o1.y = acc[5] + x1.y + bv;
    o1.z = acc[6] + x1.z + bv;  o1.w = acc[7] + x1.w + bv;
    float4* of = (float4*)(out + (size_t)ch * SEQ + tb);
    of[0] = o0;  of[1] = o1;
  }
}

extern "C" void kernel_launch(void* const* d_in, const int* in_sizes, int n_in,
                              void* d_out, int out_size, void* d_ws, size_t ws_size,
                              hipStream_t stream) {
  const float* x  = (const float*)d_in[0];   // (4, 2048, 2048) f32
  const float* k  = (const float*)d_in[1];   // (4, 2048, 2048) f32
  const float* b  = (const float*)d_in[2];   // (2048,) f32
  float* out = (float*)d_out;                // (4, 2048, 2048) f32

  const int channels = in_sizes[0] / SEQ;    // 8192 (batch*dim rows)
  const int blocks   = channels / 2;         // 2 channels per workgroup
  hyena_conv_wmma_bf16<<<blocks, 256, 0, stream>>>(x, k, b, out);
}
